// Attention_29171417874506
// MI455X (gfx1250) — matrix operand order repaired and checked by reading, not compiled, not run on hardware
//
#include <hip/hip_runtime.h>
#include <hip/hip_bf16.h>

typedef __attribute__((ext_vector_type(16))) __bf16 v16bf;
typedef __attribute__((ext_vector_type(8)))  float  v8f;

#define CDIM 512
#define HW   256
#define NB   100
#define WAY  5
#define SHOT 5
#define NQ   75
#define LSTRIDE 40   // 32 bf16 K-values + 8 pad (80 bytes/row, 16B-aligned)

__device__ __forceinline__ unsigned short f2bf(float f) {
    unsigned u = __float_as_uint(f);
    u += 0x7FFFu + ((u >> 16) & 1u);          // round-to-nearest-even
    return (unsigned short)(u >> 16);
}
__device__ __forceinline__ unsigned sortable(float f) {
    unsigned u = __float_as_uint(f);
    return (u & 0x80000000u) ? ~u : (u | 0x80000000u);
}
__device__ __forceinline__ float unsortable(unsigned u) {
    return (u & 0x80000000u) ? __uint_as_float(u ^ 0x80000000u)
                             : __uint_as_float(~u);
}

union Frag16 { v16bf v; uint4 q[2]; };

// ---------------------------------------------------------------------------
// K0: init sortable max buffer
__global__ void fsl_init(unsigned* maxb) {
    if (threadIdx.x < 128) maxb[threadIdx.x] = 0u;   // 0 sorts below sortable(-inf)
}

// ---------------------------------------------------------------------------
// K1: per-sample GEMM  logits[b] = W @ x[b]  (bf16 WMMA, f32 accum)
//     grid (8, 100): x = m-tile(0..3) | n-tile(0..1)<<2 ; 256 thr = 8 waves
__global__ __launch_bounds__(256)
void fsl_gemm(const float* __restrict__ x, const float* __restrict__ Wm,
              const float* __restrict__ bias, float* __restrict__ L,
              unsigned* __restrict__ maxb) {
    __shared__ unsigned short lA[128 * LSTRIDE];   // W rows,  K-contiguous
    __shared__ unsigned short lB[128 * LSTRIDE];   // x cols,  K-contiguous
    __shared__ float red[256];

    const int b      = blockIdx.y;
    const int tile_m = (blockIdx.x & 3) * 128;
    const int tile_n = (blockIdx.x >> 2) * 128;
    const int tid    = threadIdx.x;
    const int lane   = tid & 31;
    const int wave   = tid >> 5;
    const int wm     = wave >> 1;      // 0..3  -> 32 rows of M
    const int wn     = wave & 1;       // 0..1  -> 64 cols of N

    const float* xb = x + (size_t)b * (CDIM * HW);

    v8f acc[2][4];
    v8f z = {0.f,0.f,0.f,0.f,0.f,0.f,0.f,0.f};
    #pragma unroll
    for (int i = 0; i < 2; i++)
        #pragma unroll
        for (int j = 0; j < 4; j++) acc[i][j] = z;

    const int tk    = tid & 7;      // W loader: 4 consecutive k
    const int tmr   = tid >> 3;     // W loader: 32 rows / pass
    const int nloc  = tid & 127;    // x loader: one n column (LDS row)
    const int khalf = tid >> 7;     // x loader: K half (0: k0-15, 1: k16-31)

    for (int kk = 0; kk < CDIM; kk += 32) {
        // --- stage W tile 128x32 (fp32 -> bf16), packed uint2 DS stores ---
        #pragma unroll
        for (int i = 0; i < 4; i++) {
            int row = tmr + i * 32;
            const float4 wv = *(const float4*)&Wm[(size_t)(tile_m + row) * CDIM + kk + tk * 4];
            union { unsigned short h[4]; uint2 q; } pk;
            pk.h[0] = f2bf(wv.x); pk.h[1] = f2bf(wv.y);
            pk.h[2] = f2bf(wv.z); pk.h[3] = f2bf(wv.w);
            *(uint2*)&lA[row * LSTRIDE + tk * 4] = pk.q;
        }
        // --- stage x tile 32x128 transposed: thread owns (n, K-half),
        //     16 coalesced b32 loads -> two packed ds_store_b128 ---
        {
            union { unsigned short h[16]; uint4 q[2]; } pk;
            #pragma unroll
            for (int i = 0; i < 16; i++) {
                pk.h[i] = f2bf(xb[(size_t)(kk + khalf * 16 + i) * HW + tile_n + nloc]);
            }
            *(uint4*)&lB[nloc * LSTRIDE + khalf * 16]     = pk.q[0];
            *(uint4*)&lB[nloc * LSTRIDE + khalf * 16 + 8] = pk.q[1];
        }
        __syncthreads();

        // --- fragments per ISA 16-bit A/B layouts ---
        Frag16 af[2], bf[4];
        const int hl  = lane & 15;
        const int aks = (lane < 16) ? 0 : 8;    // A: K 0-7/16-23 vs 8-15/24-31
        #pragma unroll
        for (int fm = 0; fm < 2; fm++) {
            const unsigned short* p = &lA[(wm * 32 + fm * 16 + hl) * LSTRIDE];
            af[fm].q[0] = *(const uint4*)&p[aks];
            af[fm].q[1] = *(const uint4*)&p[aks + 16];
        }
        const int bks = (lane < 16) ? 0 : 8;    // B: K 0-7/16-23 vs 8-15/24-31
        #pragma unroll
        for (int fn = 0; fn < 4; fn++) {
            const unsigned short* p = &lB[(wn * 64 + fn * 16 + hl) * LSTRIDE];
            bf[fn].q[0] = *(const uint4*)&p[bks];
            bf[fn].q[1] = *(const uint4*)&p[bks + 16];
        }
        #pragma unroll
        for (int fm = 0; fm < 2; fm++)
            #pragma unroll
            for (int fn = 0; fn < 4; fn++)
                acc[fm][fn] = __builtin_amdgcn_wmma_f32_16x16x32_bf16(
                    false, af[fm].v, false, bf[fn].v,
                    (short)0, acc[fm][fn], false, false);
        __syncthreads();
    }

    // --- epilogue: logits + bias, block max -> sortable atomic max ---
    float lmax = -3.0e38f;
    const int rofs = (lane >= 16) ? 8 : 0;
    float* Lb = L + (size_t)b * (CDIM * HW);
    #pragma unroll
    for (int fm = 0; fm < 2; fm++) {
        int mb = tile_m + wm * 32 + fm * 16 + rofs;
        #pragma unroll
        for (int fn = 0; fn < 4; fn++) {
            int nb = tile_n + wn * 64 + fn * 16 + (lane & 15);
            #pragma unroll
            for (int r = 0; r < 8; r++) {
                float v = acc[fm][fn][r] + bias[mb + r];
                Lb[(size_t)(mb + r) * HW + nb] = v;
                lmax = fmaxf(lmax, v);
            }
        }
    }
    red[tid] = lmax; __syncthreads();
    for (int o = 128; o > 0; o >>= 1) {
        if (tid < o) red[tid] = fmaxf(red[tid], red[tid + o]);
        __syncthreads();
    }
    if (tid == 0) atomicMax(&maxb[b], sortable(red[0]));
}

// ---------------------------------------------------------------------------
// K2: per-sample sum of exp(logit - max)
__global__ __launch_bounds__(256)
void fsl_sumexp(const float* __restrict__ L, const unsigned* __restrict__ maxb,
                float* __restrict__ sums) {
    const int b = blockIdx.x;
    const float mx = unsortable(maxb[b]);
    const float* p = L + (size_t)b * (CDIM * HW);
    float s = 0.f;
    for (int i = threadIdx.x; i < CDIM * HW; i += 256) s += __expf(p[i] - mx);
    __shared__ float red[256];
    red[threadIdx.x] = s; __syncthreads();
    for (int o = 128; o > 0; o >>= 1) {
        if (threadIdx.x < o) red[threadIdx.x] += red[threadIdx.x + o];
        __syncthreads();
    }
    if (threadIdx.x == 0) sums[b] = red[0];
}

// ---------------------------------------------------------------------------
// K3: x' = x * softmax(logits), in place over L  (float4, 12800 blocks)
__global__ __launch_bounds__(256)
void fsl_apply(const float* __restrict__ x, float* __restrict__ L,
               const unsigned* __restrict__ maxb, const float* __restrict__ sums) {
    const size_t i = ((size_t)blockIdx.x * 256 + threadIdx.x) * 4;
    const int b = (int)(i >> 17);                  // 131072 = 1<<17 per sample
    const float mx  = unsortable(maxb[b]);
    const float inv = 1.0f / sums[b];
    const float4 xv = *(const float4*)&x[i];
    float4 lv = *(const float4*)&L[i];
    lv.x = xv.x * __expf(lv.x - mx) * inv;
    lv.y = xv.y * __expf(lv.y - mx) * inv;
    lv.z = xv.z * __expf(lv.z - mx) * inv;
    lv.w = xv.w * __expf(lv.w - mx) * inv;
    *(float4*)&L[i] = lv;
}

// ---------------------------------------------------------------------------
// K4: prototypes  s[way][c] = mean over 5 shots x 256 f   (2560 blocks)
__global__ __launch_bounds__(256)
void fsl_proto(const float* __restrict__ L, float* __restrict__ proto) {
    const int way = blockIdx.x >> 9;
    const int c   = blockIdx.x & 511;
    float s = 0.f;
    #pragma unroll
    for (int sh = 0; sh < SHOT; sh++) {
        const int b = way * 20 + sh;
        s += L[((size_t)b * CDIM + c) * HW + threadIdx.x];
    }
    __shared__ float red[256];
    red[threadIdx.x] = s; __syncthreads();
    for (int o = 128; o > 0; o >>= 1) {
        if (threadIdx.x < o) red[threadIdx.x] += red[threadIdx.x + o];
        __syncthreads();
    }
    if (threadIdx.x == 0) proto[way * CDIM + c] = red[0] * (1.0f / (SHOT * HW));
}

// ---------------------------------------------------------------------------
// K5: query norms  qn[n][f] = max(sqrt(sum_c q^2), eps)   (75 blocks)
__global__ __launch_bounds__(256)
void fsl_qnorm(const float* __restrict__ L, float* __restrict__ qn) {
    const int n = blockIdx.x;
    const int b = (n / 15) * 20 + SHOT + (n % 15);
    const float* q = L + (size_t)b * (CDIM * HW);
    float s = 0.f;
    for (int c = 0; c < CDIM; c++) {
        float v = q[(size_t)c * HW + threadIdx.x];
        s += v * v;
    }
    qn[n * HW + threadIdx.x] = fmaxf(sqrtf(s), 1e-8f);
}

// ---------------------------------------------------------------------------
// K6: cosine scores, class softmax, spatial mean -> out (75,5)
__global__ __launch_bounds__(256)
void fsl_score(const float* __restrict__ L, const float* __restrict__ proto,
               const float* __restrict__ qn, float* __restrict__ out) {
    __shared__ float sp[WAY * CDIM];
    __shared__ float snv[WAY];
    __shared__ float red[WAY * 256];

    const int n   = blockIdx.x;
    const int b   = (n / 15) * 20 + SHOT + (n % 15);
    const int tid = threadIdx.x;

    for (int i = tid; i < WAY * CDIM; i += 256) sp[i] = proto[i];
    __syncthreads();
    if (tid < WAY) {
        float s = 0.f;
        for (int c = 0; c < CDIM; c++) { float v = sp[tid * CDIM + c]; s += v * v; }
        snv[tid] = fmaxf(sqrtf(s), 1e-8f);
    }
    __syncthreads();

    const float* q = L + (size_t)b * (CDIM * HW);
    float dot[WAY] = {0.f, 0.f, 0.f, 0.f, 0.f};
    for (int c = 0; c < CDIM; c++) {
        float qv = q[(size_t)c * HW + tid];
        #pragma unroll
        for (int m = 0; m < WAY; m++) dot[m] += qv * sp[m * CDIM + c];
    }
    const float qnv = qn[n * HW + tid];
    float sc[WAY], mxs = -3.0e38f;
    #pragma unroll
    for (int m = 0; m < WAY; m++) {
        sc[m] = 10.0f * dot[m] / (qnv * snv[m]);
        mxs = fmaxf(mxs, sc[m]);
    }
    float se = 0.f;
    #pragma unroll
    for (int m = 0; m < WAY; m++) { sc[m] = __expf(sc[m] - mxs); se += sc[m]; }
    const float ise = 1.0f / se;
    #pragma unroll
    for (int m = 0; m < WAY; m++) red[m * 256 + tid] = sc[m] * ise;
    __syncthreads();
    for (int o = 128; o > 0; o >>= 1) {
        if (tid < o) {
            #pragma unroll
            for (int m = 0; m < WAY; m++) red[m * 256 + tid] += red[m * 256 + tid + o];
        }
        __syncthreads();
    }
    if (tid < WAY) out[n * WAY + tid] = red[tid * 256] * (1.0f / HW);
}

// ---------------------------------------------------------------------------
extern "C" void kernel_launch(void* const* d_in, const int* in_sizes, int n_in,
                              void* d_out, int out_size, void* d_ws, size_t ws_size,
                              hipStream_t stream) {
    const float* x    = (const float*)d_in[0];   // (100,512,16,16)
    const float* Wm   = (const float*)d_in[1];   // (512,512)
    const float* bias = (const float*)d_in[2];   // (512,)
    float* out = (float*)d_out;                  // (75,5)

    char* ws = (char*)d_ws;
    const size_t nL = (size_t)NB * CDIM * HW;    // 13,107,200 floats
    float*    L     = (float*)ws;                size_t off = nL * 4;
    unsigned* maxb  = (unsigned*)(ws + off);     off += 128 * 4;
    float*    sums  = (float*)(ws + off);        off += 128 * 4;
    float*    proto = (float*)(ws + off);        off += WAY * CDIM * 4;
    float*    qn    = (float*)(ws + off);        off += NQ * HW * 4;

    fsl_init  <<<1, 128, 0, stream>>>(maxb);
    fsl_gemm  <<<dim3(8, NB), 256, 0, stream>>>(x, Wm, bias, L, maxb);
    fsl_sumexp<<<NB, 256, 0, stream>>>(L, maxb, sums);
    fsl_apply <<<(int)(nL / 4 / 256), 256, 0, stream>>>(x, L, maxb, sums);
    fsl_proto <<<WAY * CDIM, 256, 0, stream>>>(L, proto);
    fsl_qnorm <<<NQ, 256, 0, stream>>>(L, qn);
    fsl_score <<<NQ, 256, 0, stream>>>(L, proto, qn, out);
}
